// Attention_18554258719125
// MI455X (gfx1250) — matrix operand order repaired and checked by reading, not compiled, not run on hardware
//
#include <hip/hip_runtime.h>
#include <hip/hip_bf16.h>

// ---------- types ----------
typedef __attribute__((ext_vector_type(16))) __bf16    v16bf;
typedef __attribute__((ext_vector_type(8)))  float     v8f;
typedef __attribute__((ext_vector_type(4)))  float     f32x4;
typedef __attribute__((ext_vector_type(2)))  unsigned  u32x2;
typedef __attribute__((ext_vector_type(4)))  unsigned  u32x4;
typedef __attribute__((ext_vector_type(8)))  unsigned  u32x8;
typedef unsigned short ushort_t;

// ---------- helpers ----------
__device__ __forceinline__ ushort_t f2bf(float f) {
    unsigned u = __builtin_bit_cast(unsigned, f);
    u += ((u >> 16) & 1u) + 0x7FFFu;              // round-to-nearest-even
    return (ushort_t)(u >> 16);
}

__device__ __forceinline__ v16bf combine16(u32x4 lo, u32x4 hi) {
    u32x8 c;
    c[0]=lo[0]; c[1]=lo[1]; c[2]=lo[2]; c[3]=lo[3];
    c[4]=hi[0]; c[5]=hi[1]; c[6]=hi[2]; c[7]=hi[3];
    return __builtin_bit_cast(v16bf, c);
}

// A-operand (16x32 bf16): lane m=l%16 holds K = kh*8..+7 and 16+kh*8..+7
__device__ __forceinline__ v16bf load_a_tile(const ushort_t* row, int k0, int kh) {
    const ushort_t* p0 = row + k0 + kh * 8;
    const ushort_t* p1 = row + k0 + 16 + kh * 8;
    return combine16(*(const u32x4*)p0, *(const u32x4*)p1);
}

// B-operand (32x16 bf16): lane n=l%16 holds K = kh*8..+7 and 16+kh*8..+7
__device__ __forceinline__ v16bf load_b_tile(const ushort_t* row, int k0, int kh) {
    const ushort_t* p = row + k0 + kh * 8;
    return combine16(*(const u32x4*)p, *(const u32x4*)(p + 16));
}

__device__ __forceinline__ v8f wmma_bf16(v16bf a, v16bf b, v8f c) {
    return __builtin_amdgcn_wmma_f32_16x16x32_bf16(
        false, a, false, b, (short)0, c, false, false);
}

// ---- CDNA5 async global->LDS copies (ASYNCcnt path, ISA §10 / §15.18) ----
__device__ __forceinline__ unsigned lds_off(const void* p) {
    return (unsigned)(unsigned long long)p;       // low 32 bits give the LDS address
}
__device__ __forceinline__ void async_b64(unsigned loff, const void* g) {
    asm volatile("global_load_async_to_lds_b64 %0, %1, off"
                 :: "v"(loff), "v"(g) : "memory");
}
__device__ __forceinline__ void async_b128(unsigned loff, const void* g) {
    asm volatile("global_load_async_to_lds_b128 %0, %1, off"
                 :: "v"(loff), "v"(g) : "memory");
}
__device__ __forceinline__ void wait_async() {
    asm volatile("s_wait_asynccnt 0" ::: "memory");
}

__device__ __forceinline__ float redmax16(float x) {
    #pragma unroll
    for (int off = 1; off < 16; off <<= 1) x = fmaxf(x, __shfl_xor(x, off, 32));
    return x;
}
__device__ __forceinline__ float redsum16(float x) {
    #pragma unroll
    for (int off = 1; off < 16; off <<= 1) x += __shfl_xor(x, off, 32);
    return x;
}

// ---------- fp32 -> bf16 pack ----------
__global__ void attn_pack_bf16(const float* __restrict__ in,
                               ushort_t* __restrict__ out, int n4) {
    for (int i = blockIdx.x * blockDim.x + threadIdx.x; i < n4;
         i += gridDim.x * blockDim.x) {
        f32x4 x = *(const f32x4*)(in + 4 * (size_t)i);
        u32x2 p;
        p[0] = (unsigned)f2bf(x[0]) | ((unsigned)f2bf(x[1]) << 16);
        p[1] = (unsigned)f2bf(x[2]) | ((unsigned)f2bf(x[3]) << 16);
        *(u32x2*)(out + 4 * (size_t)i) = p;
    }
}

// ---------- LDS-tiled GEMM: C[m][n] = (sum_k A[m][k]*B[n][k] + bias[n])*scale ----
// Block tile 128x128, K-chunk 32, double-buffered async global->LDS staging.
// 8 waves in a 4(m) x 2(n) grid; each wave computes 32x64 (8 WMMA accumulators).
// mode 0: bf16 normal   mode 1: bf16 transposed per-1024 segment (V^T)
// mode 2: f32 normal
#define GLDA 40   // LDS row stride in ushorts: 32 data + 8 pad (80B, bank-spread)
__global__ void attn_gemm_bf16(const ushort_t* __restrict__ A,
                               const ushort_t* __restrict__ Bw,
                               const float* __restrict__ bias,
                               void* __restrict__ Cout,
                               int M, int N, int Kd, float scale, int mode) {
    __shared__ ushort_t lA[2][128 * GLDA];
    __shared__ ushort_t lB[2][128 * GLDA];

    const int tid  = threadIdx.x;
    const int lane = tid & 31, wave = tid >> 5;
    const int nlo  = lane & 15, kh = lane >> 4;
    const int mw   = wave & 3,  nw = wave >> 2;
    const int mt0  = blockIdx.x * 128, nt0 = blockIdx.y * 128;

    const int srow  = tid >> 1;       // 0..127 (staged row)
    const int shalf = tid & 1;        // 32B segment within 64B row chunk

    auto stage = [&](int buf, int k0) {
        const ushort_t* ga = A + (size_t)(mt0 + srow) * Kd + k0 + shalf * 16;
        unsigned la = lds_off(&lA[buf][srow * GLDA + shalf * 16]);
        async_b128(la, ga);
        async_b128(la + 16, (const void*)(ga + 8));
        const ushort_t* gb = Bw + (size_t)(nt0 + srow) * Kd + k0 + shalf * 16;
        unsigned lb = lds_off(&lB[buf][srow * GLDA + shalf * 16]);
        async_b128(lb, gb);
        async_b128(lb + 16, (const void*)(gb + 8));
    };

    v8f acc[2][4] = {};
    stage(0, 0);
    for (int k0 = 0, it = 0; k0 < Kd; k0 += 32, ++it) {
        const int cur = it & 1;
        wait_async();
        __syncthreads();
        if (k0 + 32 < Kd) stage(cur ^ 1, k0 + 32);
        #pragma unroll
        for (int im = 0; im < 2; ++im) {
            const ushort_t* arow = &lA[cur][(mw * 32 + im * 16 + nlo) * GLDA];
            v16bf a = load_a_tile(arow, 0, kh);
            #pragma unroll
            for (int j = 0; j < 4; ++j) {
                const ushort_t* brow = &lB[cur][(nw * 64 + j * 16 + nlo) * GLDA];
                acc[im][j] = wmma_bf16(a, load_b_tile(brow, 0, kh), acc[im][j]);
            }
        }
    }

    #pragma unroll
    for (int im = 0; im < 2; ++im) {
        const int mbase = mt0 + mw * 32 + im * 16 + kh * 8;
        #pragma unroll
        for (int j = 0; j < 4; ++j) {
            const int ncol = nt0 + nw * 64 + j * 16 + nlo;
            const float bz = bias[ncol];
            if (mode == 0) {
                ushort_t* C = (ushort_t*)Cout;
                #pragma unroll
                for (int v = 0; v < 8; ++v)
                    C[(size_t)(mbase + v) * N + ncol] =
                        f2bf((acc[im][j][v] + bz) * scale);
            } else if (mode == 1) {       // transposed: Ct[bg][ncol][m%1024]
                ushort_t tmp[8];
                #pragma unroll
                for (int v = 0; v < 8; ++v)
                    tmp[v] = f2bf((acc[im][j][v] + bz) * scale);
                int bg = mbase >> 10, kl = mbase & 1023;
                u32x4 pk;
                pk[0] = (unsigned)tmp[0] | ((unsigned)tmp[1] << 16);
                pk[1] = (unsigned)tmp[2] | ((unsigned)tmp[3] << 16);
                pk[2] = (unsigned)tmp[4] | ((unsigned)tmp[5] << 16);
                pk[3] = (unsigned)tmp[6] | ((unsigned)tmp[7] << 16);
                *(u32x4*)((ushort_t*)Cout + ((size_t)(bg * N + ncol) * 1024 + kl)) = pk;
            } else {                      // f32 normal
                float* C = (float*)Cout;
                #pragma unroll
                for (int v = 0; v < 8; ++v)
                    C[(size_t)(mbase + v) * N + ncol] = acc[im][j][v] + bz;
            }
        }
    }
}

// ---------- fused attention ----------
// Block = 8 waves = 8 q-tiles of ONE head; K/V tiles async-staged into LDS once
// per block and shared by all waves (8x less L2 traffic than per-wave loads).
// Pass 1: online softmax stats. Pass 2: recompute S, write probs (the 256MB
// HBM bound), P->bf16 via per-wave LDS into A-layout, O += P@V with WMMA.
#define LKV 72    // K stage row stride (64 data + 8 pad) ushorts
#define LVV 40    // V stage row stride (32 data + 8 pad) ushorts
__global__ void attn_fused(const ushort_t* __restrict__ Qb,
                           const ushort_t* __restrict__ Kb,
                           const ushort_t* __restrict__ Vt,
                           ushort_t* __restrict__ Rb,
                           float* __restrict__ probs) {
    __shared__ ushort_t kbuf[2][32 * LKV];
    __shared__ ushort_t vbuf[2][64 * LVV];
    __shared__ ushort_t plds[8][16][32];

    const int tid  = threadIdx.x;
    const int lane = tid & 31, wave = tid >> 5;
    const int nlo  = lane & 15, kh = lane >> 4;

    const int head = blockIdx.x >> 3;            // 0..63
    const int qgrp = blockIdx.x & 7;
    const int bg   = head >> 4, hh = head & 15;
    const int qt   = qgrp * 8 + wave;            // this wave's 16-query tile

    const ushort_t* qrow  = Qb + (size_t)(bg * 1024 + qt * 16 + nlo) * 1024 + hh * 64;
    const ushort_t* kbase = Kb + (size_t)bg * 1024 * 1024 + hh * 64;
    const ushort_t* vbase = Vt + (size_t)(bg * 1024 + hh * 64) * 1024;

    // cooperative stages (all 256 threads)
    auto stageK16 = [&](int buf, int key0) {     // 16 rows x 128B = 2KB
        int row = tid >> 4, seg = tid & 15;      // 8B per thread
        const ushort_t* g = kbase + (size_t)(key0 + row) * 1024 + seg * 4;
        async_b64(lds_off(&kbuf[buf][row * LKV + seg * 4]), g);
    };
    auto stageKV32 = [&](int buf, int key0) {    // 32x128B K + 64x64B V = 8KB
        {
            int row = tid >> 3, seg = tid & 7;   // 16B per thread (K)
            const ushort_t* g = kbase + (size_t)(key0 + row) * 1024 + seg * 8;
            async_b128(lds_off(&kbuf[buf][row * LKV + seg * 8]), g);
        }
        {
            int row = tid >> 2, seg = tid & 3;   // 16B per thread (V)
            const ushort_t* g = vbase + (size_t)row * 1024 + key0 + seg * 8;
            async_b128(lds_off(&vbuf[buf][row * LVV + seg * 8]), g);
        }
    };

    // Q A-tiles reused across all key tiles (head dim 64 = 2 chunks of K=32)
    const v16bf qa0 = load_a_tile(qrow, 0,  kh);
    const v16bf qa1 = load_a_tile(qrow, 32, kh);

    float mrun[8], lrun[8];
    #pragma unroll
    for (int v = 0; v < 8; ++v) { mrun[v] = -3.0e38f; lrun[v] = 0.f; }

    // ---- pass 1: softmax statistics ----
    stageK16(0, 0);
    for (int kt = 0; kt < 64; ++kt) {
        const int cur = kt & 1;
        wait_async();
        __syncthreads();
        if (kt + 1 < 64) stageK16(cur ^ 1, (kt + 1) * 16);

        const ushort_t* krow = &kbuf[cur][nlo * LKV];
        v8f s = {};
        s = wmma_bf16(qa0, load_b_tile(krow, 0,  kh), s);
        s = wmma_bf16(qa1, load_b_tile(krow, 32, kh), s);
        #pragma unroll
        for (int v = 0; v < 8; ++v) {
            float mnew = fmaxf(mrun[v], redmax16(s[v]));
            float esum = redsum16(__expf(s[v] - mnew));
            lrun[v] = lrun[v] * __expf(mrun[v] - mnew) + esum;
            mrun[v] = mnew;
        }
    }
    float linv[8];
    #pragma unroll
    for (int v = 0; v < 8; ++v) linv[v] = 1.0f / lrun[v];

    // ---- pass 2: probs out + O = P @ V ----
    v8f o[4] = {};
    float* prow = probs + ((size_t)(bg * 16 + hh) * 1024 + qt * 16) * 1024;

    __syncthreads();
    stageKV32(0, 0);
    for (int kp = 0; kp < 32; ++kp) {            // 32 keys per iteration
        const int cur = kp & 1;
        wait_async();
        __syncthreads();
        if (kp + 1 < 32) stageKV32(cur ^ 1, (kp + 1) * 32);

        #pragma unroll
        for (int t = 0; t < 2; ++t) {
            const int kt = kp * 2 + t;
            const ushort_t* krow = &kbuf[cur][(t * 16 + nlo) * LKV];
            v8f s = {};
            s = wmma_bf16(qa0, load_b_tile(krow, 0,  kh), s);
            s = wmma_bf16(qa1, load_b_tile(krow, 32, kh), s);
            #pragma unroll
            for (int v = 0; v < 8; ++v) {
                float p = __expf(s[v] - mrun[v]) * linv[v];
                prow[(size_t)(kh * 8 + v) * 1024 + kt * 16 + nlo] = p;
                plds[wave][kh * 8 + v][t * 16 + nlo] = f2bf(p);
            }
        }
        // same-wave LDS ops are in-order (DScnt): no barrier needed
        v16bf pa = load_a_tile(&plds[wave][nlo][0], 0, kh);
        #pragma unroll
        for (int j = 0; j < 4; ++j) {
            const ushort_t* vrow = &vbuf[cur][(j * 16 + nlo) * LVV];
            o[j] = wmma_bf16(pa, load_b_tile(vrow, 0, kh), o[j]);
        }
    }

    // ---- store R (bf16 normal layout) ----
    #pragma unroll
    for (int j = 0; j < 4; ++j)
        #pragma unroll
        for (int v = 0; v < 8; ++v) {
            size_t row = (size_t)(bg * 1024 + qt * 16 + kh * 8 + v);
            Rb[row * 1024 + hh * 64 + j * 16 + nlo] = f2bf(o[j][v]);
        }
}

// ---------- launch ----------
extern "C" void kernel_launch(void* const* d_in, const int* in_sizes, int n_in,
                              void* d_out, int out_size, void* d_ws, size_t ws_size,
                              hipStream_t stream) {
    const float* xq = (const float*)d_in[0];
    const float* xkv = (const float*)d_in[1];
    const float* Wq = (const float*)d_in[2];
    const float* bq = (const float*)d_in[3];
    const float* Wk = (const float*)d_in[4];
    const float* bk = (const float*)d_in[5];
    const float* Wv = (const float*)d_in[6];
    const float* bv = (const float*)d_in[7];
    const float* Wo = (const float*)d_in[8];
    const float* bo = (const float*)d_in[9];

    const int M = 4096, N = 1024, K = 1024;     // B*G*SEQ rows
    ushort_t* ws = (ushort_t*)d_ws;             // offsets in ushort elements
    ushort_t* XQ  = ws;                         // 4M
    ushort_t* XKV = ws + 4194304;               // 4M
    ushort_t* WQb = ws + 8388608;               // 1M each
    ushort_t* WKb = ws + 9437184;
    ushort_t* WVb = ws + 10485760;
    ushort_t* WOb = ws + 11534336;
    ushort_t* Qb  = ws + 12582912;              // 4M
    ushort_t* Kbf = ws + 16777216;              // 4M
    ushort_t* VT  = ws + 20971520;              // 4M
    ushort_t* Rb  = ws + 25165824;              // 4M   (end: 56 MB)

    attn_pack_bf16<<<4096, 256, 0, stream>>>(xq,  XQ,  4194304 / 4);
    attn_pack_bf16<<<4096, 256, 0, stream>>>(xkv, XKV, 4194304 / 4);
    attn_pack_bf16<<<1024, 256, 0, stream>>>(Wq, WQb, 1048576 / 4);
    attn_pack_bf16<<<1024, 256, 0, stream>>>(Wk, WKb, 1048576 / 4);
    attn_pack_bf16<<<1024, 256, 0, stream>>>(Wv, WVb, 1048576 / 4);
    attn_pack_bf16<<<1024, 256, 0, stream>>>(Wo, WOb, 1048576 / 4);

    dim3 ggrid(M / 128, N / 128);               // (32, 8)
    // Q = (Xq Wq^T + bq) * 1/sqrt(64)   (scale folded into Q)
    attn_gemm_bf16<<<ggrid, 256, 0, stream>>>(XQ,  WQb, bq, Qb,  M, N, K, 0.125f, 0);
    attn_gemm_bf16<<<ggrid, 256, 0, stream>>>(XKV, WKb, bk, Kbf, M, N, K, 1.0f,   0);
    attn_gemm_bf16<<<ggrid, 256, 0, stream>>>(XKV, WVb, bv, VT,  M, N, K, 1.0f,   1);

    float* out_f  = (float*)d_out;              // [4096][1024] fp32
    float* probs  = out_f + 4194304;            // [4][16][1024][1024] fp32
    attn_fused<<<512, 256, 0, stream>>>(Qb, Kbf, VT, Rb, probs);

    attn_gemm_bf16<<<ggrid, 256, 0, stream>>>(Rb, WOb, bo, out_f, M, N, K, 1.0f, 2);
}
